// ROPE_MultiheadAttention_51049981280683
// MI455X (gfx1250) — matrix operand order repaired and checked by reading, not compiled, not run on hardware
//
#include <hip/hip_runtime.h>

// ---------------------------------------------------------------------------
// Types for CDNA5 WMMA (wave32, 16x16x32 bf16 -> f32)
// ---------------------------------------------------------------------------
typedef __attribute__((ext_vector_type(16))) __bf16 v16bf;
typedef __attribute__((ext_vector_type(8)))  float  v8f;

union frag_u {
    v16bf v;
    uint4 q[2];
};

__device__ inline v8f v8f_zero() {
    v8f z;
#pragma unroll
    for (int i = 0; i < 8; ++i) z[i] = 0.0f;
    return z;
}

// ---------------------------------------------------------------------------
// CDNA5 async global->LDS copy (ASYNCcnt-tracked). The builtin's params are
// (v4i __device__*, v4i __shared__*, imm offset, imm cpol) per the compiler
// diagnostic. Falls back to a plain 16B LDS store when absent.
// ---------------------------------------------------------------------------
#if defined(__has_builtin)
#if __has_builtin(__builtin_amdgcn_global_load_async_to_lds_b128)
#define HAVE_ASYNC_LDS 1
#endif
#endif

#ifdef HAVE_ASYNC_LDS
typedef int v4i __attribute__((vector_size(16)));
typedef __attribute__((address_space(1))) v4i* gv4i_p;
typedef __attribute__((address_space(3))) v4i* lv4i_p;
#endif

__device__ inline void cp_b128(const __bf16* g, __bf16* l) {
#ifdef HAVE_ASYNC_LDS
    __builtin_amdgcn_global_load_async_to_lds_b128((gv4i_p)g, (lv4i_p)l, 0, 0);
#else
    *(uint4*)l = *(const uint4*)g;
#endif
}

__device__ inline void cp_wait() {
#ifdef HAVE_ASYNC_LDS
    asm volatile("s_wait_asynccnt 0x0" ::: "memory");
#endif
}

// ---------------------------------------------------------------------------
// Fragment loaders (CDNA5 ISA VGPR layouts, CDNA5 ISA), all 16B LDS loads:
//  A (16x32 bf16, MxK) from row-major [M][K] tile:
//    lane l: M=l%16; elems 0..7 -> K=(l/16)*8+e, elems 8..15 -> K=16+(l/16)*8+(e-8)
//  B (32x16 bf16, KxN) from K-contiguous [N][K] tile:
//    lane l: N=l%16; elems 0..7 -> K=(l/16)*8+e, elems 8..15 -> K=16+(l/16)*8+(e-8)
//  C/D (16x16 f32): lane l: N=l%16; vgpr r -> M = r + 8*(l/16)
// ---------------------------------------------------------------------------
__device__ inline v16bf frag_a(const __bf16* base, int ld) {
    int l = threadIdx.x & 31;
    int m = l & 15;
    int half = l >> 4;
    const __bf16* p = base + m * ld + half * 8;
    frag_u u;
    u.q[0] = *(const uint4*)p;
    u.q[1] = *(const uint4*)(p + 16);
    return u.v;
}

__device__ inline v16bf frag_b_nk(const __bf16* base, int ld) {
    int l = threadIdx.x & 31;
    int n = l & 15;
    int half = l >> 4;
    const __bf16* p = base + n * ld + half * 8;
    frag_u u;
    u.q[0] = *(const uint4*)p;
    u.q[1] = *(const uint4*)(p + 16);
    return u.v;
}

// ---------------------------------------------------------------------------
// fp32 -> bf16 conversion (plain, and transposing for weights)
// ---------------------------------------------------------------------------
__global__ void cvt_f32_bf16(const float* __restrict__ in, __bf16* __restrict__ out, int n) {
    int i = blockIdx.x * blockDim.x + threadIdx.x;
    if (i < n) out[i] = (__bf16)in[i];
}

// in: [K][N] fp32 row-major -> out: [N][K] bf16 (weights pre-transposed once)
__global__ void cvt_transpose_f32_bf16(const float* __restrict__ in, __bf16* __restrict__ out,
                                       int K, int N) {
    int i = blockIdx.x * blockDim.x + threadIdx.x;  // i = n*K + k (output-linear)
    if (i >= K * N) return;
    int k = i % K;
    int n = i / K;
    out[i] = (__bf16)in[(size_t)k * N + n];
}

// ---------------------------------------------------------------------------
// bf16 GEMM: C[M,N] = A[M,K] * Bᵀ-stored-B; A row-major [M][K], BT is [N][K].
// Block tile 128x128, K-step 32. 8 waves in 4(m) x 2(n); wave -> 32x64 out.
// All LDS staging is contiguous b128 (async when available).
// ---------------------------------------------------------------------------
template <bool OUT_F32>
__global__ __launch_bounds__(256)
void gemm_bf16(const __bf16* __restrict__ A, const __bf16* __restrict__ BT,
               void* __restrict__ Cout, int M, int N, int K) {
    constexpr int TM = 128, TN = 128, TK = 32;
    __shared__ __bf16 As[TM][TK];  // 8 KB, [m][k]
    __shared__ __bf16 Bt[TN][TK];  // 8 KB, [n][k]

    const int m0 = blockIdx.y * TM;
    const int n0 = blockIdx.x * TN;
    const int tid = threadIdx.x;
    const int wave = tid >> 5;
    const int wm = wave & 3;   // 0..3 -> 32 rows
    const int wn = wave >> 2;  // 0..1 -> 64 cols

    v8f acc[2][4];
#pragma unroll
    for (int i = 0; i < 2; ++i)
#pragma unroll
        for (int j = 0; j < 4; ++j) acc[i][j] = v8f_zero();

    const int r = tid >> 1;          // 0..127
    const int c = (tid & 1) << 4;    // 0 or 16

    for (int k0 = 0; k0 < K; k0 += TK) {
        const __bf16* asrc = A  + (size_t)(m0 + r) * K + k0 + c;
        const __bf16* bsrc = BT + (size_t)(n0 + r) * K + k0 + c;
        cp_b128(asrc,     &As[r][c]);
        cp_b128(asrc + 8, &As[r][c + 8]);
        cp_b128(bsrc,     &Bt[r][c]);
        cp_b128(bsrc + 8, &Bt[r][c + 8]);
        if (k0 + TK < K) {  // prefetch next tiles -> global_prefetch_b8
            __builtin_prefetch(asrc + TK, 0, 1);
            __builtin_prefetch(bsrc + TK, 0, 1);
        }
        cp_wait();
        __syncthreads();

        v16bf af[2];
#pragma unroll
        for (int ti = 0; ti < 2; ++ti)
            af[ti] = frag_a(&As[wm * 32 + ti * 16][0], TK);
#pragma unroll
        for (int tj = 0; tj < 4; ++tj) {
            v16bf bf = frag_b_nk(&Bt[wn * 64 + tj * 16][0], TK);
#pragma unroll
            for (int ti = 0; ti < 2; ++ti) {
                acc[ti][tj] = __builtin_amdgcn_wmma_f32_16x16x32_bf16(
                    false, af[ti], false, bf, (short)0, acc[ti][tj], false, false);
            }
        }
        __syncthreads();
    }

    const int l = tid & 31;
    const int half = l >> 4;
    const int nl = l & 15;
#pragma unroll
    for (int ti = 0; ti < 2; ++ti)
#pragma unroll
        for (int tj = 0; tj < 4; ++tj)
#pragma unroll
            for (int e = 0; e < 8; ++e) {
                int row = m0 + wm * 32 + ti * 16 + e + half * 8;
                int col = n0 + wn * 64 + tj * 16 + nl;
                if (OUT_F32)
                    ((float*)Cout)[(size_t)row * N + col] = acc[ti][tj][e];
                else
                    ((__bf16*)Cout)[(size_t)row * N + col] = (__bf16)acc[ti][tj][e];
            }
}

// ---------------------------------------------------------------------------
// RoPE + head-major reshape: in [B,S,H*dk] bf16 -> out [B,H,S,dk] bf16
// ---------------------------------------------------------------------------
__global__ void rope_reshape(const __bf16* __restrict__ in, const int* __restrict__ pos,
                             __bf16* __restrict__ out, int B_, int S, int H, int dk) {
    int idx = blockIdx.x * blockDim.x + threadIdx.x;
    int half_dk = dk >> 1;
    int total = B_ * S * H * half_dk;
    if (idx >= total) return;
    int i = idx % half_dk;
    int t = idx / half_dk;
    int h = t % H; t /= H;
    int s = t % S;
    int b = t / S;

    const __bf16* src = in + (size_t)(b * S + s) * (H * dk) + h * dk + 2 * i;
    float e = (float)src[0];
    float o = (float)src[1];
    float p = (float)pos[b * S + s];
    float inv = __powf(10000.0f, -(float)(2 * i) / (float)dk);
    float ang = p * inv;
    float sn, cs;
    __sincosf(ang, &sn, &cs);
    float re = e * cs - o * sn;
    float ro = e * sn + o * cs;
    __bf16* dst = out + ((size_t)((b * H + h) * S + s)) * dk + 2 * i;
    dst[0] = (__bf16)re;
    dst[1] = (__bf16)ro;
}

// V reshape to d-major: in [B,S,H*dk] bf16 -> out [B,H,dk,S] bf16
__global__ void reshape_v_T(const __bf16* __restrict__ in, __bf16* __restrict__ out,
                            int B_, int S, int H, int dk) {
    int idx = blockIdx.x * blockDim.x + threadIdx.x;  // output-linear: s fastest
    int total = B_ * S * H * dk;
    if (idx >= total) return;
    int s = idx % S;
    int t = idx / S;
    int d = t % dk; t /= dk;
    int h = t % H;
    int b = t / H;
    out[idx] = in[(size_t)(b * S + s) * (H * dk) + h * dk + d];
}

// ---------------------------------------------------------------------------
// Flash attention (causal, online softmax), per (b,h) and 64-row q tile.
// Q [B,H,S,dk] row-major; K [B,H,S,dk] ([key][d] tile == [N][K] for Q·Kᵀ);
// V pre-transposed [B,H,dk,S] so the P·V B tile stages as contiguous b128.
// O accumulators live in registers.
// ---------------------------------------------------------------------------
__global__ __launch_bounds__(256)
void flash_attn(const __bf16* __restrict__ Q, const __bf16* __restrict__ Kmat,
                const __bf16* __restrict__ VT, __bf16* __restrict__ Out,
                int B_, int S, int H) {
    constexpr int BQ = 64, BK = 64, DK = 128;
    __shared__ __bf16 Qs[BQ * DK];     // 16 KB, [q][d]
    __shared__ __bf16 KVs[BK * DK];    // 16 KB: K as [key][128], then V as [d][64]
    __shared__ float  Ss[BQ * BK];     // 16 KB scores; low half aliased as bf16 P
    __shared__ float  rowm[BQ], rowl[BQ], rowsc[BQ];
    __bf16* Ps = (__bf16*)Ss;

    const int bh = blockIdx.y;
    const int b = bh / H, h = bh % H;
    const int qt = blockIdx.x;
    const int q0 = qt * BQ;
    const int tid = threadIdx.x;
    const int wave = tid >> 5;
    const int l = tid & 31, half = l >> 4, nl = l & 15;
    const float scale = 0.08838834764831845f;  // 1/sqrt(128)

    const __bf16* Qb = Q    + (size_t)bh * S * DK;
    const __bf16* Kb = Kmat + (size_t)bh * S * DK;
    const __bf16* Vb = VT   + (size_t)bh * S * DK;  // [dk][S]

    // load Q tile (contiguous 8192 bf16)
#pragma unroll
    for (int c = 0; c < 4; ++c) {
        int e0 = (tid + c * 256) * 8;
        cp_b128(Qb + (size_t)q0 * DK + e0, &Qs[e0]);
    }
    if (tid < BQ) { rowm[tid] = -1e30f; rowl[tid] = 0.0f; }

    // wave owns O rows [orow0, orow0+16), cols [ocol0, ocol0+64) (4 tiles)
    const int orow0 = (wave >> 1) * 16;
    const int ocol0 = (wave & 1) * 64;
    v8f o[4];
#pragma unroll
    for (int t = 0; t < 4; ++t) o[t] = v8f_zero();

    cp_wait();
    __syncthreads();

    const int ktiles = qt + 1;  // causal
    for (int kt = 0; kt < ktiles; ++kt) {
        const int k0 = kt * BK;
        // ---- load K tile into KVs, row-major [key][128] ----
#pragma unroll
        for (int c = 0; c < 4; ++c) {
            int e0 = (tid + c * 256) * 8;
            cp_b128(Kb + (size_t)k0 * DK + e0, &KVs[e0]);
        }
        cp_wait();
        __syncthreads();

        // ---- S = scale * Q·Kᵀ  (16 tiles of 16x16, 2 per wave) ----
#pragma unroll
        for (int tt = 0; tt < 2; ++tt) {
            int t = wave * 2 + tt;
            int ti = t >> 2, tj = t & 3;
            v8f acc = v8f_zero();
#pragma unroll
            for (int kk = 0; kk < 4; ++kk) {
                v16bf a  = frag_a(&Qs[(ti * 16) * DK + kk * 32], DK);
                v16bf bb = frag_b_nk(&KVs[(tj * 16) * DK + kk * 32], DK);
                acc = __builtin_amdgcn_wmma_f32_16x16x32_bf16(
                    false, a, false, bb, (short)0, acc, false, false);
            }
#pragma unroll
            for (int e = 0; e < 8; ++e) {
                int qi = ti * 16 + e + half * 8;
                int kj = tj * 16 + nl;
                float v = acc[e] * scale;
                if (k0 + kj > q0 + qi) v = -1e30f;  // causal mask
                Ss[qi * BK + kj] = v;
            }
        }
        __syncthreads();

        // ---- online softmax: one thread per q row ----
        if (tid < BQ) {
            int row = tid;
            float mold = rowm[row];
            float tmax = -1e30f;
            for (int j = 0; j < BK; ++j) tmax = fmaxf(tmax, Ss[row * BK + j]);
            float mnew = fmaxf(mold, tmax);
            float f = __expf(mold - mnew);
            float sum = 0.0f;
            for (int j = 0; j < BK; ++j) {                  // read fp32 then overwrite
                float p = __expf(Ss[row * BK + j] - mnew);  // as bf16 (ascending j keeps
                sum += p;                                   // the alias safe)
                Ps[row * BK + j] = (__bf16)p;
            }
            rowl[row] = rowl[row] * f + sum;
            rowm[row] = mnew;
            rowsc[row] = f;
        }
        __syncthreads();

        // ---- rescale O accumulators; stage V tile [d][64] (contiguous b128) ----
#pragma unroll
        for (int e = 0; e < 8; ++e) {
            float f = rowsc[orow0 + e + half * 8];
#pragma unroll
            for (int t = 0; t < 4; ++t) o[t][e] *= f;
        }
#pragma unroll
        for (int c = 0; c < 4; ++c) {
            int e0 = (tid + c * 256) * 8;
            int d = e0 >> 6;        // 0..127
            int key0 = e0 & 63;     // 0,8,...,56
            cp_b128(Vb + (size_t)d * S + k0 + key0, &KVs[e0]);
        }
        cp_wait();
        __syncthreads();

        // ---- O += P·V (wave: 4 tiles x 2 K-steps) ----
        v16bf a0 = frag_a(&Ps[orow0 * BK + 0], BK);
        v16bf a1 = frag_a(&Ps[orow0 * BK + 32], BK);
#pragma unroll
        for (int t = 0; t < 4; ++t) {
            v16bf b0 = frag_b_nk(&KVs[(ocol0 + t * 16) * BK + 0], BK);
            o[t] = __builtin_amdgcn_wmma_f32_16x16x32_bf16(
                false, a0, false, b0, (short)0, o[t], false, false);
            v16bf b1 = frag_b_nk(&KVs[(ocol0 + t * 16) * BK + 32], BK);
            o[t] = __builtin_amdgcn_wmma_f32_16x16x32_bf16(
                false, a1, false, b1, (short)0, o[t], false, false);
        }
        __syncthreads();  // before KVs/Ss are overwritten next iteration
    }

    // ---- normalize and write Out [B,S,H*dk] (bf16) ----
    const int D = H * DK;
#pragma unroll
    for (int e = 0; e < 8; ++e) {
        int qi = orow0 + e + half * 8;
        float inv = 1.0f / rowl[qi];
#pragma unroll
        for (int t = 0; t < 4; ++t) {
            int col = ocol0 + t * 16 + nl;
            Out[(size_t)(b * S + (q0 + qi)) * D + h * DK + col] = (__bf16)(o[t][e] * inv);
        }
    }
}

// ---------------------------------------------------------------------------
// Launch: x -> bf16; W* -> bf16 TRANSPOSED [N][K]; QKV GEMMs; RoPE/reshape
// (V transposed to [B,H,dk,S]); flash attention; output projection (fp32).
// ---------------------------------------------------------------------------
extern "C" void kernel_launch(void* const* d_in, const int* in_sizes, int n_in,
                              void* d_out, int out_size, void* d_ws, size_t ws_size,
                              hipStream_t stream) {
    const int Bb = 2, S = 2048, D = 2048, H = 16, dk = 128;
    const float* x   = (const float*)d_in[0];
    const int*   pos = (const int*)d_in[1];
    const float* WQ  = (const float*)d_in[2];
    const float* WK  = (const float*)d_in[3];
    const float* WV  = (const float*)d_in[4];
    const float* WO  = (const float*)d_in[5];

    const size_t nx = (size_t)Bb * S * D;  // 8M elems
    const size_t nw = (size_t)D * D;       // 4M elems

    char* p = (char*)d_ws;
    __bf16* xb   = (__bf16*)p; p += nx * 2;
    __bf16* wqT  = (__bf16*)p; p += nw * 2;
    __bf16* wkT  = (__bf16*)p; p += nw * 2;
    __bf16* wvT  = (__bf16*)p; p += nw * 2;
    __bf16* woT  = (__bf16*)p; p += nw * 2;
    __bf16* qraw = (__bf16*)p; p += nx * 2;
    __bf16* kraw = (__bf16*)p; p += nx * 2;
    __bf16* vraw = (__bf16*)p; p += nx * 2;
    __bf16* qr   = (__bf16*)p; p += nx * 2;
    __bf16* kr   = (__bf16*)p; p += nx * 2;
    __bf16* vT   = (__bf16*)p; p += nx * 2;
    __bf16* attn = qraw;  // qraw is dead after rope_reshape -> reuse

    // 1) fp32 -> bf16 (x plain; weights transposed to [N][K])
    cvt_f32_bf16<<<(int)((nx + 255) / 256), 256, 0, stream>>>(x, xb, (int)nx);
    cvt_transpose_f32_bf16<<<(int)((nw + 255) / 256), 256, 0, stream>>>(WQ, wqT, D, D);
    cvt_transpose_f32_bf16<<<(int)((nw + 255) / 256), 256, 0, stream>>>(WK, wkT, D, D);
    cvt_transpose_f32_bf16<<<(int)((nw + 255) / 256), 256, 0, stream>>>(WV, wvT, D, D);
    cvt_transpose_f32_bf16<<<(int)((nw + 255) / 256), 256, 0, stream>>>(WO, woT, D, D);

    // 2) Q/K/V projections: [4096,2048] x [2048,2048]
    dim3 gg(D / 128, (Bb * S) / 128);
    gemm_bf16<false><<<gg, 256, 0, stream>>>(xb, wqT, qraw, Bb * S, D, D);
    gemm_bf16<false><<<gg, 256, 0, stream>>>(xb, wkT, kraw, Bb * S, D, D);
    gemm_bf16<false><<<gg, 256, 0, stream>>>(xb, wvT, vraw, Bb * S, D, D);

    // 3) RoPE (Q,K) + head-major reshape; V transposed to [B,H,dk,S]
    int pairs = Bb * S * H * (dk / 2);
    rope_reshape<<<(pairs + 255) / 256, 256, 0, stream>>>(qraw, pos, qr, Bb, S, H, dk);
    rope_reshape<<<(pairs + 255) / 256, 256, 0, stream>>>(kraw, pos, kr, Bb, S, H, dk);
    reshape_v_T<<<(int)((nx + 255) / 256), 256, 0, stream>>>(vraw, vT, Bb, S, H, dk);

    // 4) causal flash attention -> attn [B,S,D] bf16
    dim3 ga(S / 64, Bb * H);
    flash_attn<<<ga, 256, 0, stream>>>(qr, kr, vT, attn, Bb, S, H);

    // 5) output projection -> fp32 d_out
    gemm_bf16<true><<<gg, 256, 0, stream>>>(attn, woT, d_out, Bb * S, D, D);
}
